// AdaptiveSSM2DRefiner_2173253452149
// MI455X (gfx1250) — matrix operand order repaired and checked by reading, not compiled, not run on hardware
//
#include <hip/hip_runtime.h>
#include <hip/hip_bf16.h>

// ---------------------------------------------------------------------------
// AdaptiveSSM2DRefiner (S5 block) for MI455X / gfx1250, wave32 + WMMA bf16.
//   B=16, C=384, H=W=64  -> N = 65536 tokens, L = 4096, P = 192 (2P = 384)
// Pipeline:
//   prep (ZOH disc, bandlimit, bf16 weights) -> LN1 -> GEMM1(Bu)
//   -> chunked scan (3 phases) -> GEMM2(+D*fx, GELU, residual)
//   -> LN2 -> GEGLU-GEMM -> GEMM4(+residual, NCHW store)
// GEMMs: v_wmma_f32_16x16x32_bf16, f32 accumulate. B-slab staged in LDS once
// per block (kills 8x redundant B loads + per-WMMA L2 stalls); A fragments
// are pipelined global loads.
// ---------------------------------------------------------------------------

typedef __attribute__((ext_vector_type(16))) __bf16 v16bf;
typedef __attribute__((ext_vector_type(8)))  __bf16 v8bf;
typedef __attribute__((ext_vector_type(8)))  float  v8f;

#define NTOK   65536
#define CD     384
#define PP     192
#define TWOP   384
#define LL     4096
#define BB     16
#define CHN    16      // scan chunks per sequence
#define SS     256     // chunk length (LL / CHN)
#define TWOC   768

// ---- workspace layout (bytes, every region 256B aligned) -------------------
#define SZ_FX32   ((size_t)NTOK * CD * 4)      // fx f32; later y (in place)
#define SZ_BU     ((size_t)NTOK * TWOP * 4)    // Bu / local scan; later fy f32
#define SZ_FXBF   ((size_t)NTOK * CD * 2)      // fx bf16; later fy bf16
#define SZ_XSBF   ((size_t)NTOK * TWOP * 2)    // xs bf16; later geglu-out bf16
#define SZ_CARRY  ((size_t)BB * CHN * PP * 4)  // 196608
#define SZ_SMALL  ((size_t)1024)               // per-P vectors, padded

#define OFF_FX32  ((size_t)0)
#define OFF_BU    (OFF_FX32 + SZ_FX32)
#define OFF_FXBF  (OFF_BU   + SZ_BU)
#define OFF_XSBF  (OFF_FXBF + SZ_FXBF)
#define OFF_ENDR  (OFF_XSBF + SZ_XSBF)
#define OFF_ENDI  (OFF_ENDR + SZ_CARRY)
#define OFF_PRER  (OFF_ENDI + SZ_CARRY)
#define OFF_PREI  (OFF_PRER + SZ_CARRY)
#define OFF_LAMR  (OFF_PREI + SZ_CARRY)
#define OFF_LAMI  (OFF_LAMR + SZ_SMALL)
#define OFF_ASR   (OFF_LAMI + SZ_SMALL)
#define OFF_ASI   (OFF_ASR  + SZ_SMALL)
#define OFF_COEFR (OFF_ASI  + SZ_SMALL)
#define OFF_COEFI (OFF_COEFR + SZ_SMALL)
#define OFF_MASK  (OFF_COEFI + SZ_SMALL)
#define OFF_W1    (OFF_MASK + SZ_SMALL)                    // [2P, C] bf16
#define OFF_CCAT  (OFF_W1   + (size_t)TWOP * CD * 2)       // [C, 2P] bf16
#define OFF_WENC  (OFF_CCAT + (size_t)CD * TWOP * 2)       // [2C, C] bf16
#define OFF_WDEC  (OFF_WENC + (size_t)TWOC * CD * 2)       // [C, C]  bf16

// ---------------------------------------------------------------------------
__device__ __forceinline__ unsigned short f2bf(float f) {
  unsigned int u = __float_as_uint(f);
  u += 0x7FFFu + ((u >> 16) & 1u);          // round-to-nearest-even
  return (unsigned short)(u >> 16);
}

__device__ __forceinline__ float gelu_exact(float x) {
  return 0.5f * x * (1.0f + erff(x * 0.70710678118654752f));
}

// A fragment: 16x32 bf16. Per-lane base is loop-invariant:
//   arow = A + (m0 + lane%16)*lda + 16*(lane/16)
// lane<16 holds K in {k0+0..15}; lane>=16 shifted by +16.
__device__ __forceinline__ const __bf16* a_base(const unsigned short* A,
                                                int lda, int m0, int lane) {
  return reinterpret_cast<const __bf16*>(A) +
         (size_t)(m0 + (lane & 15)) * lda + 16 * ((lane >> 4) & 1);
}
__device__ __forceinline__ v16bf frag_a(const __bf16* arow, int k0) {
  v8bf lo = *reinterpret_cast<const v8bf*>(arow + k0);
  v8bf up = *reinterpret_cast<const v8bf*>(arow + k0 + 8);
  return __builtin_shufflevector(lo, up, 0, 1, 2, 3, 4, 5, 6, 7,
                                 8, 9, 10, 11, 12, 13, 14, 15);
}

// B fragment from the LDS-staged slab, layout [col][K] with K contiguous:
//   bcol = lds + (ncol_local + lane%16)*CD + 16*(lane/16)
__device__ __forceinline__ v16bf frag_b(const unsigned short* bcol, int k0) {
  return *reinterpret_cast<const v16bf*>(
      reinterpret_cast<const __bf16*>(bcol) + k0);
}

#define WMMA_BF16(acc, a, b) \
  __builtin_amdgcn_wmma_f32_16x16x32_bf16(false, (a), false, (b), (short)0, \
                                          (acc), false, false)

// Cooperative stage of `cols` contiguous weight rows (K-major) into LDS.
__device__ __forceinline__ void stage_slab(unsigned short* lds,
                                           const unsigned short* src,
                                           int elems, int tid) {
  for (int idx = tid * 8; idx < elems; idx += 256 * 8)
    *reinterpret_cast<uint4*>(lds + idx) =
        *reinterpret_cast<const uint4*>(src + idx);
}

// Double-buffered 16x64 tile GEMM core, K = 384 (12 steps of 32).
// B fragments come from LDS (ldsB holds 64 cols x 384 K, K contiguous).
__device__ __forceinline__ void gemm_tile4(const unsigned short* A, int lda,
                                           const unsigned short* ldsB,
                                           int m0, int lane, v8f acc[4]) {
  const __bf16* arow = a_base(A, lda, m0, lane);
  int cl = lane & 15, bh = 16 * ((lane >> 4) & 1);
  const unsigned short* bc0 = ldsB + (size_t)(cl +  0) * CD + bh;
  const unsigned short* bc1 = ldsB + (size_t)(cl + 16) * CD + bh;
  const unsigned short* bc2 = ldsB + (size_t)(cl + 32) * CD + bh;
  const unsigned short* bc3 = ldsB + (size_t)(cl + 48) * CD + bh;
  v16bf a = frag_a(arow, 0);
  #pragma unroll
  for (int i = 0; i < CD / 32; ++i) {
    const int k0 = i * 32, kn = k0 + 32;
    v16bf an;
    if (kn < CD) an = frag_a(arow, kn);     // keep next A load in flight
    acc[0] = WMMA_BF16(acc[0], a, frag_b(bc0, k0));
    acc[1] = WMMA_BF16(acc[1], a, frag_b(bc1, k0));
    acc[2] = WMMA_BF16(acc[2], a, frag_b(bc2, k0));
    acc[3] = WMMA_BF16(acc[3], a, frag_b(bc3, k0));
    if (kn < CD) a = an;
  }
}

// ---------------------------------------------------------------------------
// prep: discretization constants per state p (ZOH), a^S, bandlimit mask
__global__ void kprep_small(const float* Lambda, const float* log_step,
                            float* lamre, float* lamim, float* aSre, float* aSim,
                            float* coefre, float* coefim, float* maskv) {
  int p = threadIdx.x;
  if (p >= PP) return;
  float lr = Lambda[2 * p], li = Lambda[2 * p + 1];
  float st = expf(log_step[p]);                       // STEP_SCALE = 1
  float er = expf(lr * st);
  float lbr = er * cosf(li * st), lbi = er * sinf(li * st);
  lamre[p] = lbr; lamim[p] = lbi;
  float wr = lbr, wi = lbi;                           // a^256 by 8 squarings
  #pragma unroll
  for (int i = 0; i < 8; ++i) { float nr = wr*wr - wi*wi, ni = 2.f*wr*wi; wr = nr; wi = ni; }
  aSre[p] = wr; aSim[p] = wi;
  float nr = lbr - 1.0f, ni = lbi;                    // (lam_bar-1)/lam
  float den = lr * lr + li * li;
  coefre[p] = (nr * lr + ni * li) / den;
  coefim[p] = (ni * lr - nr * li) / den;
  maskv[p] = (st * fabsf(li) * (1.0f / 6.28318530718f) < 0.25f) ? 1.0f : 0.0f;
}

// W1[2P, C] = [Re(B_bar); Im(B_bar)] in bf16
__global__ void kprep_w1(const float* Bmat, const float* coefre,
                         const float* coefim, unsigned short* W1) {
  int i = blockIdx.x * 256 + threadIdx.x;            // i in [0, P*C)
  if (i >= PP * CD) return;
  int p = i / CD, h = i % CD;
  float br = Bmat[(size_t)i * 2], bi = Bmat[(size_t)i * 2 + 1];
  float cr = coefre[p], ci = coefim[p];
  W1[(size_t)p * CD + h]          = f2bf(cr * br - ci * bi);
  W1[(size_t)(PP + p) * CD + h]   = f2bf(cr * bi + ci * br);
}

// Ccat[C, 2P]: cols [0,P) = 2*mask*Re(C), cols [P,2P) = -2*mask*Im(C)
__global__ void kprep_ccat(const float* Cmat, const float* maskv,
                           unsigned short* Ccat) {
  int i = blockIdx.x * 256 + threadIdx.x;            // i in [0, C*P)
  if (i >= CD * PP) return;
  int h = i / PP, p = i % PP;
  float m = maskv[p];
  Ccat[(size_t)h * TWOP + p]       = f2bf(2.0f * m * Cmat[(size_t)i * 2]);
  Ccat[(size_t)h * TWOP + PP + p]  = f2bf(-2.0f * m * Cmat[(size_t)i * 2 + 1]);
}

__global__ void kconv_bf16(const float* in, unsigned short* out, int n) {
  int i = blockIdx.x * 256 + threadIdx.x;
  if (i < n) out[i] = f2bf(in[i]);
}

// ---------------------------------------------------------------------------
// LN1: input x[B,C,L] (NCHW); one thread per token; coalesced x reads.
__global__ void kln1(const float* __restrict__ x, const float* g,
                     const float* bta, float* fx32, unsigned short* fxbf) {
  int t = blockIdx.x * 256 + threadIdx.x;
  int b = t >> 12, l = t & (LL - 1);
  const float* base = x + (size_t)b * CD * LL + l;
  float s = 0.f, ss = 0.f;
  for (int c = 0; c < CD; ++c) { float v = base[(size_t)c * LL]; s += v; ss += v * v; }
  float mu = s * (1.0f / CD);
  float inv = rsqrtf(ss * (1.0f / CD) - mu * mu + 1e-5f);
  for (int c = 0; c < CD; ++c) {
    float v = base[(size_t)c * LL];
    float o = (v - mu) * inv * g[c] + bta[c];
    fx32[(size_t)t * CD + c] = o;
    fxbf[(size_t)t * CD + c] = f2bf(o);
  }
}

// LN2: input y[N,C] row-major; one wave per token.
__global__ void kln2(const float* __restrict__ y, const float* g,
                     const float* bta, float* fy32, unsigned short* fybf) {
  int lane = threadIdx.x & 31, wv = threadIdx.x >> 5;
  int t = blockIdx.x * 8 + wv;
  const float* row = y + (size_t)t * CD;
  float s = 0.f, ss = 0.f;
  for (int c = lane; c < CD; c += 32) { float v = row[c]; s += v; ss += v * v; }
  #pragma unroll
  for (int o = 16; o > 0; o >>= 1) { s += __shfl_xor(s, o, 32); ss += __shfl_xor(ss, o, 32); }
  float mu = s * (1.0f / CD);
  float inv = rsqrtf(ss * (1.0f / CD) - mu * mu + 1e-5f);
  for (int c = lane; c < CD; c += 32) {
    float o = (row[c] - mu) * inv * g[c] + bta[c];
    fy32[(size_t)t * CD + c] = o;
    fybf[(size_t)t * CD + c] = f2bf(o);
  }
}

// ---------------------------------------------------------------------------
// GEMM1: Bu[N,2P] = fx_bf[N,C] @ W1[2P,C]^T, f32 store.
__global__ __launch_bounds__(256)
void kgemm_bu(const unsigned short* __restrict__ A,
              const unsigned short* __restrict__ W, float* __restrict__ Cout) {
  __shared__ __align__(16) unsigned short ldsB[64 * CD];
  stage_slab(ldsB, W + (size_t)blockIdx.y * 64 * CD, 64 * CD, threadIdx.x);
  __syncthreads();
  int lane = threadIdx.x & 31, wv = threadIdx.x >> 5;
  int m0 = blockIdx.x * 128 + wv * 16;
  int n0 = blockIdx.y * 64;
  v8f acc[4] = {{}, {}, {}, {}};
  gemm_tile4(A, CD, ldsB, m0, lane, acc);
  int hi = (lane >> 4) & 1, cl = lane & 15;
  #pragma unroll
  for (int r = 0; r < 8; ++r) {
    float* p = Cout + (size_t)(m0 + r + 8 * hi) * TWOP + n0 + cl;
    p[0] = acc[0][r]; p[16] = acc[1][r]; p[32] = acc[2][r]; p[48] = acc[3][r];
  }
}

// GEMM2: ylin = xs_bf[N,2P] @ Ccat[C,2P]^T + D*fx ; y = gelu(ylin)+fx,
// written in place over fx32.
__global__ __launch_bounds__(256)
void kgemm_ssm(const unsigned short* __restrict__ A,
               const unsigned short* __restrict__ W,
               const float* __restrict__ Dv, float* __restrict__ Y) {
  __shared__ __align__(16) unsigned short ldsB[64 * CD];
  stage_slab(ldsB, W + (size_t)blockIdx.y * 64 * TWOP, 64 * TWOP, threadIdx.x);
  __syncthreads();
  int lane = threadIdx.x & 31, wv = threadIdx.x >> 5;
  int m0 = blockIdx.x * 128 + wv * 16;
  int n0 = blockIdx.y * 64;
  v8f acc[4] = {{}, {}, {}, {}};
  gemm_tile4(A, TWOP, ldsB, m0, lane, acc);
  int hi = (lane >> 4) & 1, cl = lane & 15;
  #pragma unroll
  for (int r = 0; r < 8; ++r) {
    int row = m0 + r + 8 * hi;
    float* p = Y + (size_t)row * CD + n0 + cl;
    #pragma unroll
    for (int j = 0; j < 4; ++j) {
      int col = n0 + cl + 16 * j;
      float fx = p[16 * j];
      float yl = acc[j][r] + Dv[col] * fx;
      p[16 * j] = gelu_exact(yl) + fx;
    }
  }
}

// GEGLU GEMM: g[N,C] = z1 * gelu(z2); z = fy_bf @ W_enc[2C,C]^T.
// Each wave computes the matching z1 (rows n) and z2 (rows C+n) tiles;
// both weight slabs staged in LDS.
__global__ __launch_bounds__(256)
void kgemm_geglu(const unsigned short* __restrict__ A,
                 const unsigned short* __restrict__ W,
                 unsigned short* __restrict__ G) {
  __shared__ __align__(16) unsigned short ldsB[2 * 64 * CD];
  int n0 = blockIdx.y * 64;
  stage_slab(ldsB,           W + (size_t)n0 * CD,        64 * CD, threadIdx.x);
  stage_slab(ldsB + 64 * CD, W + (size_t)(CD + n0) * CD, 64 * CD, threadIdx.x);
  __syncthreads();
  int lane = threadIdx.x & 31, wv = threadIdx.x >> 5;
  int m0 = blockIdx.x * 128 + wv * 16;
  const __bf16* arow = a_base(A, CD, m0, lane);
  int cl = lane & 15, bh = 16 * ((lane >> 4) & 1);
  const unsigned short* bc[8];
  #pragma unroll
  for (int j = 0; j < 4; ++j) {
    bc[j]     = ldsB + (size_t)(cl + 16 * j) * CD + bh;            // z1
    bc[4 + j] = ldsB + (size_t)(64 + cl + 16 * j) * CD + bh;       // z2
  }
  v8f z1[4] = {{}, {}, {}, {}}, z2[4] = {{}, {}, {}, {}};
  v16bf a = frag_a(arow, 0);
  #pragma unroll
  for (int i = 0; i < CD / 32; ++i) {
    const int k0 = i * 32, kn = k0 + 32;
    v16bf an;
    if (kn < CD) an = frag_a(arow, kn);
    #pragma unroll
    for (int j = 0; j < 4; ++j) {
      z1[j] = WMMA_BF16(z1[j], a, frag_b(bc[j], k0));
      z2[j] = WMMA_BF16(z2[j], a, frag_b(bc[4 + j], k0));
    }
    if (kn < CD) a = an;
  }
  int hi = (lane >> 4) & 1;
  #pragma unroll
  for (int r = 0; r < 8; ++r) {
    int row = m0 + r + 8 * hi;
    unsigned short* p = G + (size_t)row * CD + n0 + cl;
    #pragma unroll
    for (int j = 0; j < 4; ++j)
      p[16 * j] = f2bf(z1[j][r] * gelu_exact(z2[j][r]));
  }
}

// GEMM4: out = g_bf @ W_dec[C,C]^T + fy ; stored transposed to [B,C,H,W].
__global__ __launch_bounds__(256)
void kgemm_out(const unsigned short* __restrict__ A,
               const unsigned short* __restrict__ W,
               const float* __restrict__ FY, float* __restrict__ Out) {
  __shared__ __align__(16) unsigned short ldsB[64 * CD];
  stage_slab(ldsB, W + (size_t)blockIdx.y * 64 * CD, 64 * CD, threadIdx.x);
  __syncthreads();
  int lane = threadIdx.x & 31, wv = threadIdx.x >> 5;
  int m0 = blockIdx.x * 128 + wv * 16;
  int n0 = blockIdx.y * 64;
  v8f acc[4] = {{}, {}, {}, {}};
  gemm_tile4(A, CD, ldsB, m0, lane, acc);
  int hi = (lane >> 4) & 1, cl = lane & 15;
  #pragma unroll
  for (int r = 0; r < 8; ++r) {
    int row = m0 + r + 8 * hi;             // token t
    int b = row >> 12, l = row & (LL - 1);
    const float* fy = FY + (size_t)row * CD + n0 + cl;
    #pragma unroll
    for (int j = 0; j < 4; ++j) {
      int col = n0 + cl + 16 * j;
      Out[(size_t)b * CD * LL + (size_t)col * LL + l] = acc[j][r] + fy[16 * j];
    }
  }
}

// ---------------------------------------------------------------------------
// Scan phase A: local scan of each 256-chunk, in place over Bu; emit carries.
__global__ void kscanA(float* __restrict__ Bu, const float* lamre,
                       const float* lamim, float* endre, float* endim) {
  int bc = blockIdx.x;                 // b*CHN + chunk
  int b = bc / CHN, ch = bc % CHN, p = threadIdx.x;
  float ar = lamre[p], ai = lamim[p];
  float* base = Bu + ((size_t)b * LL + (size_t)ch * SS) * TWOP;
  float xr = 0.f, xi = 0.f;
  for (int i = 0; i < SS; ++i) {
    float br = base[(size_t)i * TWOP + p];
    float bi = base[(size_t)i * TWOP + PP + p];
    float nr = ar * xr - ai * xi + br;
    float ni = ar * xi + ai * xr + bi;
    base[(size_t)i * TWOP + p] = nr;
    base[(size_t)i * TWOP + PP + p] = ni;
    xr = nr; xi = ni;
  }
  endre[(size_t)bc * PP + p] = xr;
  endim[(size_t)bc * PP + p] = xi;
}

// Phase B: serial combine of the 16 chunk carries per (b,p); emit prefixes.
__global__ void kscanB(const float* endre, const float* endim,
                       const float* aSre, const float* aSim,
                       float* prere, float* preim) {
  int b = blockIdx.x, p = threadIdx.x;
  float asr = aSre[p], asi = aSim[p];
  float cr = 0.f, ci = 0.f;
  for (int ch = 0; ch < CHN; ++ch) {
    size_t idx = ((size_t)b * CHN + ch) * PP + p;
    prere[idx] = cr; preim[idx] = ci;
    float er = endre[idx], ei = endim[idx];
    float nr = er + asr * cr - asi * ci;
    float ni = ei + asr * ci + asi * cr;
    cr = nr; ci = ni;
  }
}

// Phase C: xs[l] = local[l] + a^(l-l0+1)*prefix; emit bf16 [N,2P].
__global__ void kscanC(const float* __restrict__ Bu, const float* lamre,
                       const float* lamim, const float* prere,
                       const float* preim, unsigned short* __restrict__ xsbf) {
  int bc = blockIdx.x;
  int b = bc / CHN, ch = bc % CHN, p = threadIdx.x;
  float ar = lamre[p], ai = lamim[p];
  float pr = prere[(size_t)bc * PP + p], pi = preim[(size_t)bc * PP + p];
  size_t t0 = (size_t)b * LL + (size_t)ch * SS;
  const float* base = Bu + t0 * TWOP;
  unsigned short* ob = xsbf + t0 * TWOP;
  float wr = ar, wi = ai;
  for (int i = 0; i < SS; ++i) {
    float lr = base[(size_t)i * TWOP + p];
    float li = base[(size_t)i * TWOP + PP + p];
    ob[(size_t)i * TWOP + p]      = f2bf(lr + wr * pr - wi * pi);
    ob[(size_t)i * TWOP + PP + p] = f2bf(li + wr * pi + wi * pr);
    float nr = wr * ar - wi * ai, ni = wr * ai + wi * ar;
    wr = nr; wi = ni;
  }
}

// ---------------------------------------------------------------------------
extern "C" void kernel_launch(void* const* d_in, const int* in_sizes, int n_in,
                              void* d_out, int out_size, void* d_ws,
                              size_t ws_size, hipStream_t stream) {
  const float* x        = (const float*)d_in[0];
  const float* ln1_g    = (const float*)d_in[1];
  const float* ln1_b    = (const float*)d_in[2];
  const float* Lambda   = (const float*)d_in[3];
  const float* Bmat     = (const float*)d_in[4];
  const float* Cmat     = (const float*)d_in[5];
  const float* Dv       = (const float*)d_in[6];
  const float* log_step = (const float*)d_in[7];
  const float* ln2_g    = (const float*)d_in[8];
  const float* ln2_b    = (const float*)d_in[9];
  const float* W_enc    = (const float*)d_in[10];
  const float* W_dec    = (const float*)d_in[11];
  float* out = (float*)d_out;

  char* ws = (char*)d_ws;
  float* fx32  = (float*)(ws + OFF_FX32);           // fx; then y (in place)
  float* Bu    = (float*)(ws + OFF_BU);             // Bu/local-scan; then fy32
  unsigned short* fxbf = (unsigned short*)(ws + OFF_FXBF);  // fx bf; then fy bf
  unsigned short* xsbf = (unsigned short*)(ws + OFF_XSBF);  // xs bf; then g bf
  float* endre = (float*)(ws + OFF_ENDR);
  float* endim = (float*)(ws + OFF_ENDI);
  float* prere = (float*)(ws + OFF_PRER);
  float* preim = (float*)(ws + OFF_PREI);
  float* lamre = (float*)(ws + OFF_LAMR);
  float* lamim = (float*)(ws + OFF_LAMI);
  float* aSre  = (float*)(ws + OFF_ASR);
  float* aSim  = (float*)(ws + OFF_ASI);
  float* coefre = (float*)(ws + OFF_COEFR);
  float* coefim = (float*)(ws + OFF_COEFI);
  float* maskv  = (float*)(ws + OFF_MASK);
  unsigned short* W1bf   = (unsigned short*)(ws + OFF_W1);
  unsigned short* Ccatbf = (unsigned short*)(ws + OFF_CCAT);
  unsigned short* Wencbf = (unsigned short*)(ws + OFF_WENC);
  unsigned short* Wdecbf = (unsigned short*)(ws + OFF_WDEC);

  // -- weight prep --
  kprep_small<<<1, 192, 0, stream>>>(Lambda, log_step, lamre, lamim, aSre, aSim,
                                     coefre, coefim, maskv);
  kprep_w1<<<(PP * CD + 255) / 256, 256, 0, stream>>>(Bmat, coefre, coefim, W1bf);
  kprep_ccat<<<(CD * PP + 255) / 256, 256, 0, stream>>>(Cmat, maskv, Ccatbf);
  kconv_bf16<<<(TWOC * CD + 255) / 256, 256, 0, stream>>>(W_enc, Wencbf, TWOC * CD);
  kconv_bf16<<<(CD * CD + 255) / 256, 256, 0, stream>>>(W_dec, Wdecbf, CD * CD);

  // -- LN1 --
  kln1<<<NTOK / 256, 256, 0, stream>>>(x, ln1_g, ln1_b, fx32, fxbf);

  // -- GEMM1: Bu = fx @ B_bar^T (stacked re/im) --
  kgemm_bu<<<dim3(NTOK / 128, TWOP / 64), 256, 0, stream>>>(fxbf, W1bf, Bu);

  // -- chunked linear-recurrence scan --
  kscanA<<<BB * CHN, PP, 0, stream>>>(Bu, lamre, lamim, endre, endim);
  kscanB<<<BB, PP, 0, stream>>>(endre, endim, aSre, aSim, prere, preim);
  kscanC<<<BB * CHN, PP, 0, stream>>>(Bu, lamre, lamim, prere, preim, xsbf);

  // -- GEMM2: y = gelu(2Re(C xs) + D*fx) + fx, in place over fx32 --
  kgemm_ssm<<<dim3(NTOK / 128, CD / 64), 256, 0, stream>>>(xsbf, Ccatbf, Dv, fx32);

  // -- LN2 (fy32 -> Bu buffer, fy bf16 -> fxbf buffer) --
  kln2<<<NTOK / 8, 256, 0, stream>>>(fx32, ln2_g, ln2_b, Bu, fxbf);

  // -- GEGLU GEMM (g bf16 -> xsbf buffer) --
  kgemm_geglu<<<dim3(NTOK / 128, CD / 64), 256, 0, stream>>>(fxbf, Wencbf, xsbf);

  // -- final GEMM + residual + NCHW store --
  kgemm_out<<<dim3(NTOK / 128, CD / 64), 256, 0, stream>>>(xsbf, Wdecbf, Bu, out);
}
